// FFQNet_81527069212716
// MI455X (gfx1250) — matrix operand order repaired and checked by reading, not compiled, not run on hardware
//
#include <hip/hip_runtime.h>
#include <hip/hip_bf16.h>
#include <stdint.h>

typedef __attribute__((ext_vector_type(16))) __bf16 v16bf;
typedef __attribute__((ext_vector_type(8)))  float  v8f;
typedef __attribute__((ext_vector_type(4)))  unsigned int u32x4;
typedef __attribute__((ext_vector_type(4)))  float  f32x4;
typedef uint16_t u16;
typedef uint32_t u32;
typedef uint64_t u64;

// Problem dims (fixed by the reference)
constexpr int B_ROWS = 65536;
constexpr int D      = 1024;
constexpr int H1     = 2048;
constexpr int H2     = 1024;
constexpr int AOUT   = 64;
constexpr float EPS  = 1e-8f;

// Tiling
constexpr int MT = 32;               // rows per workgroup
constexpr int S1 = D  + 8;           // bf16 LDS row stride (x / h2 tiles)
constexpr int S2 = H1 + 8;           // bf16 LDS row stride (h1 tile)
constexpr int XS_BYTES  = MT * S1 * 2;                        // 66,048
constexpr int H1S_BYTES = MT * S2 * 2;                        // 131,584
constexpr int LDS_BYTES = XS_BYTES + H1S_BYTES + 2 * MT * 4;  // 197,888

// ---- explicit global-address-space accessors: force global_load/store,
// ---- keep weight streaming off the flat path (flat couples LOADcnt+DScnt)
#define AS1 __attribute__((address_space(1)))
__device__ __forceinline__ u32x4 gload_u4(const void* p) { return *(AS1 const u32x4*)p; }
__device__ __forceinline__ f32x4 gload_f4(const void* p) { return *(AS1 const f32x4*)p; }
__device__ __forceinline__ float gload_f (const void* p) { return *(AS1 const float*)p; }
__device__ __forceinline__ void  gstore_f(void* p, float v) { *(AS1 float*)p = v; }
__device__ __forceinline__ void  gstore_u64(void* p, u64 v) { *(AS1 u64*)p = v; }

__device__ __forceinline__ u16 f32_to_bf16(float f) {
  u32 u = __float_as_uint(f);
  u32 r = u + 0x7FFFu + ((u >> 16) & 1u);   // round-to-nearest-even
  return (u16)(r >> 16);
}

__device__ __forceinline__ u64 pack4_bf16(f32x4 v) {
  return (u64)f32_to_bf16(v.x)
       | ((u64)f32_to_bf16(v.y) << 16)
       | ((u64)f32_to_bf16(v.z) << 32)
       | ((u64)f32_to_bf16(v.w) << 48);
}

union BF16x16 { u32x4 q[2]; v16bf v; };

// ---------------------------------------------------------------------------
// Weight pre-conversion: fp32 row-major [N,K] -> bf16 row-major [N,K] in d_ws
// ---------------------------------------------------------------------------
__global__ void convert_bf16_4(const float* __restrict__ s, u16* __restrict__ d, int n4) {
  int i = blockIdx.x * blockDim.x + threadIdx.x;
  if (i < n4) {
    f32x4 v = gload_f4((const char*)s + (size_t)i * 16);
    gstore_u64((u64*)d + i, pack4_bf16(v));
  }
}

// ---------------------------------------------------------------------------
// Dual-row-subtile micro-kernel: one wave computes (2 x NT) 16x16 D-tiles
// covering rows 0..31 of the stripe. Each B tile (2 global b128/lane) feeds
// TWO WMMAs -> 1 global b128 per WMMA, and no cross-wave weight duplication.
// ---------------------------------------------------------------------------
template<int NT>
__device__ __forceinline__ void gemm2r(const u16* aLds, int aStride, int K,
                                       const u16* __restrict__ wG,
                                       int nBase, int lane,
                                       v8f* acc0 /*rows 0-15*/, v8f* acc1 /*rows 16-31*/) {
  const int lh = lane >> 4;       // K-half selector (ISA A/B operand layouts)
  const int ll = lane & 15;       // M index (A) / N index (B)
  const u16* ap0 = aLds + ll * aStride + (lh << 3);
  const u16* ap1 = aLds + (16 + ll) * aStride + (lh << 3);
  const u16* bp[NT];
#pragma unroll
  for (int t = 0; t < NT; ++t)
    bp[t] = wG + (size_t)(nBase + (t << 4) + ll) * K + (lh << 3);

  for (int k = 0; k < K; k += 32) {
    BF16x16 a0, a1;
    a0.q[0] = *(const u32x4*)(ap0);       // K {0..7}/{8..15} by lane half
    a0.q[1] = *(const u32x4*)(ap0 + 16);  // K {16..23}/{24..31}
    a1.q[0] = *(const u32x4*)(ap1);
    a1.q[1] = *(const u32x4*)(ap1 + 16);
#pragma unroll
    for (int t = 0; t < NT; ++t) {
      BF16x16 b;
      b.q[0] = gload_u4(bp[t]);
      b.q[1] = gload_u4(bp[t] + 16);
      acc0[t] = __builtin_amdgcn_wmma_f32_16x16x32_bf16(
          false, a0.v, false, b.v, (short)0, acc0[t], false, false);
      acc1[t] = __builtin_amdgcn_wmma_f32_16x16x32_bf16(
          false, a1.v, false, b.v, (short)0, acc1[t], false, false);
      bp[t] += 32;
    }
    ap0 += 32;
    ap1 += 32;
  }
}

// Single-subtile variant (layer 3: 8 waves <-> 8 output tiles)
__device__ __forceinline__ void gemm1(const u16* aLds, int aStride, int K,
                                      const u16* __restrict__ wG,
                                      int rowSub, int nBase, int lane, v8f& acc) {
  const int lh = lane >> 4, ll = lane & 15;
  const u16* ap = aLds + (rowSub + ll) * aStride + (lh << 3);
  const u16* bp = wG + (size_t)(nBase + ll) * K + (lh << 3);
  for (int k = 0; k < K; k += 32) {
    BF16x16 a, b;
    a.q[0] = *(const u32x4*)(ap);
    a.q[1] = *(const u32x4*)(ap + 16);
    b.q[0] = gload_u4(bp);
    b.q[1] = gload_u4(bp + 16);
    acc = __builtin_amdgcn_wmma_f32_16x16x32_bf16(
        false, a.v, false, b.v, (short)0, acc, false, false);
    ap += 32;
    bp += 32;
  }
}

// Epilogue: y = relu(scale_row * acc + bias_col) -> bf16 LDS tile; optional
// per-row sum(y^2) accumulation (shfl_xor within 16-lane halves + ds_add_f32).
template<int NT, bool SUMSQ>
__device__ __forceinline__ void epilogue_hidden(v8f* acc,
                                                const float* scl,
                                                const float* __restrict__ bias,
                                                int rowSub, int nBase, int lane,
                                                u16* dLds, int dStride, float* sumNew) {
  const int lh = lane >> 4, ll = lane & 15;
  float bb[NT];
#pragma unroll
  for (int t = 0; t < NT; ++t) bb[t] = gload_f(bias + nBase + (t << 4) + ll);
#pragma unroll
  for (int r = 0; r < 8; ++r) {
    const int m = rowSub + (lh << 3) + r;     // C/D layout: VGPR r + lane half -> row
    const float s = scl[m];
    float p = 0.f;
#pragma unroll
    for (int t = 0; t < NT; ++t) {
      float v = fmaxf(acc[t][r] * s + bb[t], 0.f);
      p += v * v;
      dLds[m * dStride + nBase + (t << 4) + ll] = f32_to_bf16(v);
    }
    if constexpr (SUMSQ) {
      p += __shfl_xor(p, 1, 32);
      p += __shfl_xor(p, 2, 32);
      p += __shfl_xor(p, 4, 32);
      p += __shfl_xor(p, 8, 32);
      if (ll == 0) atomicAdd(&sumNew[m], p);  // ds_add_f32
    }
  }
}

// ---------------------------------------------------------------------------
// Fused kernel: one 32-row stripe per workgroup, 8 waves.
// Layers 1/2: wave w owns column group w (1/8 of N), both row subtiles.
// ---------------------------------------------------------------------------
__global__ __launch_bounds__(256)
void ffq_fused(const float* __restrict__ states,
               const float* __restrict__ b1, const float* __restrict__ b2,
               const float* __restrict__ bq,
               const u16* __restrict__ W1b, const u16* __restrict__ W2b,
               const u16* __restrict__ Wqb,
               float* __restrict__ out) {
  extern __shared__ char lds[];
  u16*   xs   = (u16*)lds;                      // 32 x S1 bf16 (x tile, later h2 tile)
  u16*   h1s  = (u16*)(lds + XS_BYTES);         // 32 x S2 bf16
  u16*   h2s  = xs;                             // alias: xs dead once layer-1 GEMMs finish
  float* sum  = (float*)(lds + XS_BYTES + H1S_BYTES);
  float* scl  = sum + MT;

  const int tid     = threadIdx.x;
  const int lane    = tid & 31;
  const int w       = tid >> 5;
  const int rowBase = blockIdx.x * MT;

  if (tid < MT) sum[tid] = 0.f;
  __syncthreads();

  // ---- Phase 0: stage x stripe -> bf16 LDS, fp32 row sum-of-squares ----
  {
    const int row  = tid >> 3;       // 8 threads per row
    const int part = tid & 7;        // 128 contiguous floats each
    const float* src = states + (size_t)(rowBase + row) * D + part * 128;
    u16* dst = xs + row * S1 + part * 128;
    float p = 0.f;
#pragma unroll 4
    for (int j = 0; j < 128; j += 4) {
      f32x4 v = gload_f4(src + j);
      p += v.x * v.x + v.y * v.y + v.z * v.z + v.w * v.w;
      *(u64*)(dst + j) = pack4_bf16(v);
    }
    atomicAdd(&sum[row], p);
  }
  __syncthreads();
  if (tid < MT) { scl[tid] = 1.f / (sqrtf(sum[tid]) + EPS); sum[tid] = 0.f; }
  __syncthreads();

  // ---- Layer 1: h1 = relu(s_x * (x . W1^T) + b1), 32x2048, K=1024 ----
  // wave w -> N in [w*256, w*256+256), 4 passes of 64 cols, 2x4 tiles each
  for (int p = 0; p < 4; ++p) {
    const int nB = w * 256 + p * 64;
    v8f a0[4] = {}, a1[4] = {};
    gemm2r<4>(xs, S1, D, W1b, nB, lane, a0, a1);
    epilogue_hidden<4, true>(a0, scl, b1, 0,  nB, lane, h1s, S2, sum);
    epilogue_hidden<4, true>(a1, scl, b1, 16, nB, lane, h1s, S2, sum);
  }
  __syncthreads();
  if (tid < MT) scl[tid] = 1.f / (sqrtf(sum[tid]) + EPS);
  __syncthreads();

  // ---- Layer 2: h2 = relu(s_h1 * (h1 . W2^T) + b2), 32x1024, K=2048 ----
  // wave w -> N in [w*128, w*128+128), 2 passes of 64 cols
  for (int p = 0; p < 2; ++p) {
    const int nB = w * 128 + p * 64;
    v8f a0[4] = {}, a1[4] = {};
    gemm2r<4>(h1s, S2, H1, W2b, nB, lane, a0, a1);
    epilogue_hidden<4, false>(a0, scl, b2, 0,  nB, lane, h2s, S1, nullptr);
    epilogue_hidden<4, false>(a1, scl, b2, 16, nB, lane, h2s, S1, nullptr);
  }
  __syncthreads();

  // ---- Layer 3: q = h2 . Wq^T + bq, 32x64, K=1024 (one tile per wave) ----
  {
    const int rowSub = (w >> 2) << 4;
    const int nB     = (w & 3) << 4;
    v8f acc = {};
    gemm1(h2s, S1, H2, Wqb, rowSub, nB, lane, acc);
    const int lh = lane >> 4, ll = lane & 15;
    const float b = gload_f(bq + nB + ll);
#pragma unroll
    for (int r = 0; r < 8; ++r) {
      const int m = rowSub + (lh << 3) + r;
      gstore_f(out + (size_t)(rowBase + m) * AOUT + nB + ll, acc[r] + b);
    }
  }
}

// ---------------------------------------------------------------------------
extern "C" void kernel_launch(void* const* d_in, const int* in_sizes, int n_in,
                              void* d_out, int out_size, void* d_ws, size_t ws_size,
                              hipStream_t stream) {
  const float* states = (const float*)d_in[0];
  const float* W1     = (const float*)d_in[1];
  const float* b1     = (const float*)d_in[2];
  const float* W2     = (const float*)d_in[3];
  const float* b2     = (const float*)d_in[4];
  const float* Wq     = (const float*)d_in[5];
  const float* bq     = (const float*)d_in[6];
  float* out          = (float*)d_out;

  // bf16 weight mirrors in workspace (8.5 MB; resident in 192 MB L2)
  u16* W1b = (u16*)d_ws;
  u16* W2b = W1b + (size_t)H1 * D;
  u16* Wqb = W2b + (size_t)H2 * H1;

  const int n1 = H1 * D / 4, n2 = H2 * H1 / 4, n3 = AOUT * H2 / 4;
  convert_bf16_4<<<(n1 + 255) / 256, 256, 0, stream>>>(W1, W1b, n1);
  convert_bf16_4<<<(n2 + 255) / 256, 256, 0, stream>>>(W2, W2b, n2);
  convert_bf16_4<<<(n3 + 255) / 256, 256, 0, stream>>>(Wq, Wqb, n3);

  (void)hipFuncSetAttribute((const void*)ffq_fused,
                            hipFuncAttributeMaxDynamicSharedMemorySize, LDS_BYTES);
  ffq_fused<<<B_ROWS / MT, 256, LDS_BYTES, stream>>>(states, b1, b2, bq,
                                                     W1b, W2b, Wqb, out);
}
